// MambaBlock_48069273977390
// MI455X (gfx1250) — matrix operand order repaired and checked by reading, not compiled, not run on hardware
//
#include <hip/hip_runtime.h>

// ---- problem sizes (fixed by reference) ----
#define B_   2
#define L_   2048
#define DM   1024
#define DI   2048
#define NS   16
#define KC   4
#define RR   64
#define ROWS (B_ * L_)        // 4096
#define DXZ  (2 * DI)         // 4096
#define DBCW (RR + 2 * NS)    // 96

// ---- WMMA types (gfx1250, wave32) ----
typedef __attribute__((ext_vector_type(16))) __bf16 v16bf;
typedef __attribute__((ext_vector_type(8)))  float  v8f;

union FragBF { v16bf v; unsigned short s[16]; uint4 q[2]; };

__device__ __forceinline__ unsigned short f2bf(float f) {
  union { float f; unsigned int u; } c; c.f = f;
  unsigned int u = c.u;
  u += 0x7fffu + ((u >> 16) & 1u);   // round-to-nearest-even
  return (unsigned short)(u >> 16);
}

__device__ __forceinline__ float silu_f(float x) {
  return x / (1.f + __expf(-x));
}

// async VMEM -> LDS copy of one 16B chunk (ASYNCcnt-tracked, no VGPR round-trip)
__device__ __forceinline__ void async_cp16(unsigned ldsAddr, const void* gaddr) {
  asm volatile("global_load_async_to_lds_b128 %0, %1, off"
               :: "v"(ldsAddr), "v"((unsigned long long)gaddr) : "memory");
}
// low 32 bits of a generic LDS pointer == LDS byte offset (aperture is in high half)
__device__ __forceinline__ unsigned lds_addr(const void* p) {
  return (unsigned)(unsigned long long)p;
}

// =====================================================================
// GEMM: C[M,N] = A[M,K] * Bt[N,K]^T   (A,Bt bf16 in global, fp32 accum)
// block = 256 threads = 8 wave32s; block tile 128x128; wave tile 32x64
// -> 8 v_wmma_f32_16x16x32_bf16 per wave per K-step.
// Double-buffered LDS fed by GLOBAL_LOAD_ASYNC_TO_LDS_B128; software
// pipeline is PEELED so the steady-state body is branch-free:
//   issue(next) ; s_wait_asynccnt 4 ; barrier ; compute(cur) ; barrier
// mode 1: C = softplus(C + bias[col]).  Optional bf16 mirror store Cb.
// Requires: M % 128 == 0, K % 32 == 0, Bt padded so N-tiles are full.
// =====================================================================
#define BM 128
#define BN 128
#define BK 32
#define APITCH 40          // ushorts per tile row (80B: async chunks stay 16B-aligned)
#define MATELEMS (128 * APITCH)       // 5120 ushorts per matrix tile
#define BUFELEMS (2 * MATELEMS)       // A + B per pipeline slot

__global__ __launch_bounds__(256)
void gemm_async(const unsigned short* __restrict__ Ab, int lda,
                const unsigned short* __restrict__ Bt,
                float* __restrict__ C, int ldc,
                unsigned short* __restrict__ Cb,
                int M, int N, int K,
                const float* __restrict__ bias, int mode)
{
  __shared__ unsigned short smem[2 * BUFELEMS];   // 40 KB, double buffered

  const int tid  = threadIdx.x;
  const int lane = tid & 31;
  const int wave = tid >> 5;
  const int wm   = (wave >> 1) * 32;   // wave M offset (0..96)
  const int wn   = (wave & 1)  * 64;   // wave N offset (0/64)
  const int bm0  = blockIdx.y * BM;
  const int bn0  = blockIdx.x * BN;

  // staging geometry: 128 rows x 32 k per matrix = 512 x 16B chunks;
  // each thread owns 2 A-chunks + 2 B-chunks (rows srow and srow+64).
  const int srow = tid >> 2;          // 0..63
  const int scol = (tid & 3) * 8;     // k element offset within tile
  const unsigned short* gA0 = Ab + (size_t)(bm0 + srow) * lda + scol;
  const unsigned short* gA1 = gA0 + (size_t)64 * lda;
  const unsigned short* gB0 = Bt + (size_t)(bn0 + srow) * K + scol;
  const unsigned short* gB1 = gB0 + (size_t)64 * K;
  const unsigned lA0 = lds_addr(&smem[srow * APITCH + scol]);
  const unsigned lA1 = lA0 + 64 * APITCH * 2;
  const unsigned lB0 = lA0 + MATELEMS * 2;
  const unsigned lB1 = lB0 + 64 * APITCH * 2;

  auto issue = [&](int k0, int pb) {
    const unsigned bo = pb ? (unsigned)(BUFELEMS * 2) : 0u;
    async_cp16(lA0 + bo, gA0 + k0);
    async_cp16(lA1 + bo, gA1 + k0);
    async_cp16(lB0 + bo, gB0 + k0);
    async_cp16(lB1 + bo, gB1 + k0);
  };

  v8f acc[2][4] = {};
  const int mfrag = lane & 15;
  const int kbA   = (lane >> 4) * 8;    // A-frag K octet select (ISA 16x32 layout)
  const int kbB   = (lane >> 4) * 8;    // B-frag K octet select (ISA 32x16 layout)

  auto compute = [&](int pb) __attribute__((always_inline)) {
    const unsigned short* lA = &smem[pb * BUFELEMS];
    const unsigned short* lB = lA + MATELEMS;
    FragBF a[2], b[4];
    #pragma unroll
    for (int i = 0; i < 2; ++i) {       // A fragment: 2x ds_load_b128
      int m = wm + i * 16 + mfrag;
      a[i].q[0] = *(const uint4*)&lA[m * APITCH + kbA];
      a[i].q[1] = *(const uint4*)&lA[m * APITCH + 16 + kbA];
    }
    #pragma unroll
    for (int j = 0; j < 4; ++j) {       // B fragment: 2x ds_load_b128
      int n = wn + j * 16 + mfrag;
      const uint4* p = (const uint4*)&lB[n * APITCH + kbB];
      b[j].q[0] = p[0]; b[j].q[1] = p[2];
    }
    #pragma unroll
    for (int i = 0; i < 2; ++i)
      #pragma unroll
      for (int j = 0; j < 4; ++j)
        acc[i][j] = __builtin_amdgcn_wmma_f32_16x16x32_bf16(
            false, a[i].v, false, b[j].v, (short)0, acc[i][j], false, false);
  };

  issue(0, 0);                          // prologue

  // steady state: branch-free pipelined body
  int pb = 0;
  int k0 = 0;
  for (; k0 + BK < K; k0 += BK, pb ^= 1) {
    issue(k0 + BK, pb ^ 1);                           // overlap next tile
    asm volatile("s_wait_asynccnt 0x4" ::: "memory"); // current tile landed
    __syncthreads();
    compute(pb);
    __syncthreads();                                  // buffer pb free for re-issue
  }
  // epilogue: last tile
  asm volatile("s_wait_asynccnt 0x0" ::: "memory");
  __syncthreads();
  compute(pb);

  // C/D layout: VGPR v -> M = v + 8*(lane>=16), N = lane&15
  const int rbase = (lane >> 4) * 8;
  #pragma unroll
  for (int i = 0; i < 2; ++i) {
    #pragma unroll
    for (int j = 0; j < 4; ++j) {
      int col = bn0 + wn + j * 16 + (lane & 15);
      if (col < N) {
        #pragma unroll
        for (int v = 0; v < 8; ++v) {
          int row = bm0 + wm + i * 16 + rbase + v;
          float val = acc[i][j][v];
          if (mode == 1) {
            val += bias[col];
            val = (val > 20.f) ? val : logf(1.f + __expf(val));  // softplus
          }
          C[(size_t)row * ldc + col] = val;
          if (Cb) Cb[(size_t)row * ldc + col] = f2bf(val);
        }
      }
    }
  }
}

// =====================================================================
// fp32 -> bf16 elementwise (prepass for x)
// =====================================================================
__global__ __launch_bounds__(256)
void cvt_bf16(const float* __restrict__ in, unsigned short* __restrict__ out, int n)
{
  int i = blockIdx.x * 256 + threadIdx.x;
  if (i < n) out[i] = f2bf(in[i]);
}

// =====================================================================
// Transpose + convert:  in fp32 [K][Nsrc]  ->  out bf16 [Npad][K]
// (weights become N-major so GEMM B tiles are contiguous K runs).
// LDS-tiled 32x32; blockDim 256 (32x8). K%32==0, Npad%32==0.
// =====================================================================
__global__ __launch_bounds__(256)
void transpose_cvt(const float* __restrict__ in, unsigned short* __restrict__ out,
                   int K, int Nsrc, int Npad)
{
  __shared__ float t[32][33];
  int kt = blockIdx.x * 32, nt = blockIdx.y * 32;
  int tx = threadIdx.x & 31, ty = threadIdx.x >> 5;
  #pragma unroll
  for (int r = 0; r < 32; r += 8) {
    int k = kt + ty + r, n = nt + tx;
    t[ty + r][tx] = (n < Nsrc) ? in[(size_t)k * Nsrc + n] : 0.f;
  }
  __syncthreads();
  #pragma unroll
  for (int r = 0; r < 32; r += 8) {
    int n = nt + ty + r, k = kt + tx;
    if (n < Npad) out[(size_t)n * K + k] = f2bf(t[tx][ty + r]);
  }
}

// =====================================================================
// Depthwise causal conv (K=4) + SiLU; writes fp32 (for scan) + bf16 (GEMM A)
// =====================================================================
__global__ __launch_bounds__(256)
void conv_silu(const float* __restrict__ xz, const float* __restrict__ w_conv,
               const float* __restrict__ b_conv,
               float* __restrict__ uc, unsigned short* __restrict__ ucb)
{
  int idx = blockIdx.x * 256 + threadIdx.x;  // over ROWS*DI
  int d   = idx & (DI - 1);
  int row = idx >> 11;
  int l   = row & (L_ - 1);
  float acc = b_conv[d];
  #pragma unroll
  for (int k = 0; k < KC; ++k) {
    int lp = l - (KC - 1) + k;
    if (lp >= 0)
      acc += xz[(size_t)(row - (KC - 1) + k) * DXZ + d] * w_conv[d * KC + k];
  }
  float s = silu_f(acc);
  uc[idx]  = s;
  ucb[idx] = f2bf(s);
}

// =====================================================================
// Selective scan: one thread per (b,d); 16 states in registers.
// Fuses y = (scan + u*D) * silu(z); emits bf16 directly for the out-GEMM.
// =====================================================================
__global__ __launch_bounds__(256)
void scan_kernel(const float* __restrict__ delta, const float* __restrict__ uc,
                 const float* __restrict__ dbc, const float* __restrict__ xz,
                 const float* __restrict__ A_log, const float* __restrict__ Dvec,
                 unsigned short* __restrict__ ybf)
{
  int d = blockIdx.x * 256 + threadIdx.x;
  int b = blockIdx.y;
  float h[NS], Ad[NS];
  #pragma unroll
  for (int n = 0; n < NS; ++n) { h[n] = 0.f; Ad[n] = -__expf(A_log[d * NS + n]); }
  float Dd = Dvec[d];

  for (int l = 0; l < L_; ++l) {
    int row = b * L_ + l;
    float dt = delta[(size_t)row * DI + d];
    float ut = uc[(size_t)row * DI + d];
    float du = dt * ut;
    float y = 0.f;
    #pragma unroll
    for (int n = 0; n < NS; ++n) {
      float Bv = dbc[row * DBCW + RR + n];        // block-uniform -> L2 broadcast
      float Cv = dbc[row * DBCW + RR + NS + n];
      h[n] = h[n] * __expf(dt * Ad[n]) + du * Bv;
      y += h[n] * Cv;
    }
    float zt = xz[(size_t)row * DXZ + DI + d];
    ybf[(size_t)row * DI + d] = f2bf((y + ut * Dd) * silu_f(zt));
  }
}

// =====================================================================
// LayerNorm + residual: one block per row of DM=1024.
// =====================================================================
__global__ __launch_bounds__(256)
void ln_residual(const float* __restrict__ obuf, const float* __restrict__ x,
                 const float* __restrict__ gamma, const float* __restrict__ beta,
                 float* __restrict__ out)
{
  __shared__ float ssum[256], ssq[256];
  int row = blockIdx.x;
  int tid = threadIdx.x;
  float v[4];
  float lsum = 0.f, lsq = 0.f;
  #pragma unroll
  for (int j = 0; j < 4; ++j) {
    v[j] = obuf[(size_t)row * DM + tid + j * 256];
    lsum += v[j]; lsq += v[j] * v[j];
  }
  ssum[tid] = lsum; ssq[tid] = lsq;
  __syncthreads();
  for (int s = 128; s > 0; s >>= 1) {
    if (tid < s) { ssum[tid] += ssum[tid + s]; ssq[tid] += ssq[tid + s]; }
    __syncthreads();
  }
  float mu   = ssum[0] * (1.f / DM);
  float var  = ssq[0] * (1.f / DM) - mu * mu;
  float rstd = rsqrtf(var + 1e-5f);
  #pragma unroll
  for (int j = 0; j < 4; ++j) {
    int c = tid + j * 256;
    out[(size_t)row * DM + c] =
        (v[j] - mu) * rstd * gamma[c] + beta[c] + x[(size_t)row * DM + c];
  }
}

// =====================================================================
extern "C" void kernel_launch(void* const* d_in, const int* in_sizes, int n_in,
                              void* d_out, int out_size, void* d_ws, size_t ws_size,
                              hipStream_t stream)
{
  const float* x      = (const float*)d_in[0];
  const float* W_in   = (const float*)d_in[1];
  const float* w_conv = (const float*)d_in[2];
  const float* b_conv = (const float*)d_in[3];
  const float* W_x    = (const float*)d_in[4];
  const float* W_dt   = (const float*)d_in[5];
  const float* b_dt   = (const float*)d_in[6];
  const float* A_log  = (const float*)d_in[7];
  const float* Dvec   = (const float*)d_in[8];
  const float* W_out  = (const float*)d_in[9];
  const float* gamma  = (const float*)d_in[10];
  const float* beta   = (const float*)d_in[11];
  float* out = (float*)d_out;

  // workspace layout (64B-aligned slices)
  char* w = (char*)d_ws;
  size_t off = 0;
  auto alloc = [&](size_t bytes) -> void* {
    void* p = w + off; off += (bytes + 63) & ~(size_t)63; return p;
  };
  float* xz    = (float*)alloc((size_t)ROWS * DXZ  * 4);
  float* uc    = (float*)alloc((size_t)ROWS * DI   * 4);
  float* dbc   = (float*)alloc((size_t)ROWS * DBCW * 4);
  float* delta = (float*)alloc((size_t)ROWS * DI   * 4);
  float* obuf  = (float*)alloc((size_t)ROWS * DM   * 4);
  unsigned short* xb    = (unsigned short*)alloc((size_t)ROWS * DM   * 2);
  unsigned short* ucb   = (unsigned short*)alloc((size_t)ROWS * DI   * 2);
  unsigned short* dbcb  = (unsigned short*)alloc((size_t)ROWS * DBCW * 2);
  unsigned short* ybf   = (unsigned short*)alloc((size_t)ROWS * DI   * 2);
  unsigned short* WinT  = (unsigned short*)alloc((size_t)DXZ  * DM * 2);  // [4096][1024]
  unsigned short* WxT   = (unsigned short*)alloc((size_t)128  * DI * 2);  // [128][2048] zero-padded
  unsigned short* WdtT  = (unsigned short*)alloc((size_t)DI   * RR * 2);  // [2048][64]
  unsigned short* WoutT = (unsigned short*)alloc((size_t)DM   * DI * 2);  // [1024][2048]

  dim3 blk(256);

  // ---- prepass: bf16 conversion + weight transposes ----
  cvt_bf16<<<dim3(ROWS * DM / 256), blk, 0, stream>>>(x, xb, ROWS * DM);
  transpose_cvt<<<dim3(DM / 32, DXZ / 32), blk, 0, stream>>>(W_in,  WinT,  DM, DXZ, DXZ);
  transpose_cvt<<<dim3(DI / 32, 128 / 32), blk, 0, stream>>>(W_x,   WxT,   DI, DBCW, 128);
  transpose_cvt<<<dim3(RR / 32, DI / 32),  blk, 0, stream>>>(W_dt,  WdtT,  RR, DI, DI);
  transpose_cvt<<<dim3(DI / 32, DM / 32),  blk, 0, stream>>>(W_out, WoutT, DI, DM, DM);

  // 1. xz = x @ W_in                    (4096x1024)*(1024x4096)
  gemm_async<<<dim3(DXZ / BN, ROWS / BM), blk, 0, stream>>>(
      xb, DM, WinT, xz, DXZ, nullptr, ROWS, DXZ, DM, nullptr, 0);

  // 2. u = silu(conv(u) + b)
  conv_silu<<<dim3(ROWS * DI / 256), blk, 0, stream>>>(xz, w_conv, b_conv, uc, ucb);

  // 3. dbc = u @ W_x                    (4096x2048)*(2048x96) [B padded to 128]
  gemm_async<<<dim3(1, ROWS / BM), blk, 0, stream>>>(
      ucb, DI, WxT, dbc, DBCW, dbcb, ROWS, DBCW, DI, nullptr, 0);

  // 4. delta = softplus(dt @ W_dt + b_dt)   (4096x64)*(64x2048)
  gemm_async<<<dim3(DI / BN, ROWS / BM), blk, 0, stream>>>(
      dbcb, DBCW, WdtT, delta, DI, nullptr, ROWS, DI, RR, b_dt, 1);

  // 5. scan + gating -> bf16
  scan_kernel<<<dim3(DI / 256, B_), blk, 0, stream>>>(
      delta, uc, dbc, xz, A_log, Dvec, ybf);

  // 6. out = y @ W_out                  (4096x2048)*(2048x1024)
  gemm_async<<<dim3(DM / BN, ROWS / BM), blk, 0, stream>>>(
      ybf, DI, WoutT, obuf, DM, nullptr, ROWS, DM, DI, nullptr, 0);

  // 7. LayerNorm + residual
  ln_residual<<<dim3(ROWS), blk, 0, stream>>>(obuf, x, gamma, beta, out);
}
